// GroupHeadMLP_69337952027058
// MI455X (gfx1250) — matrix operand order repaired and checked by reading, not compiled, not run on hardware
//
#include <hip/hip_runtime.h>

typedef __attribute__((ext_vector_type(16))) __bf16 v16bf;
typedef __attribute__((ext_vector_type(8)))  __bf16 v8bf;
typedef __attribute__((ext_vector_type(8)))  float  v8f;
typedef __attribute__((ext_vector_type(4)))  float  v4f;
typedef __attribute__((ext_vector_type(4)))  unsigned int v4u;
typedef int v4i_g __attribute__((vector_size(16)));   // matches builtin param type

#define NG   32
#define NF   256
#define NH   52
#define NHP  64
#define NO   5
#define NOP  16
#define NB   8192
#define NUF  512
#define NUH  103
#define NUHP 112
#define NK2P 128
#define NUO  20
#define NUOP 32
#define XSTRIDE (NG*NF + NUF)   // 8704 floats per x row
#define OUTW_N  (NG*NO + NUO)   // 180 concatenated head outputs

#define W1K (NF + 8)     // LDS row stride, K=256 (+pad vs bank conflicts)
#define W2K (NHP + 8)    // LDS row stride, K=64
#define U1K (NUF + 8)    // 520
#define U2K (NK2P + 8)   // 136

// packed bf16 weight images (exact LDS layout) in workspace, after the
// B x 180 f32 staging region
#define HWS_ELEMS ((size_t)NB * OUTW_N)                 // floats
#define GPACK   (NHP*W1K + NHP*W2K + NOP*W2K)           // 22656 bf16 / group
#define HSEG1   (NHP*W1K)                               // 16896
#define HSEG2   (NHP*W2K)                               // 4608
#define HSEG3   (NOP*W2K)                               // 1152
#define HPACK_TOT ((size_t)NG * GPACK)
#define USEG1   (NUHP*U1K)                              // 58240
#define USEG2   (NUHP*U2K)                              // 15232
#define USEG3   (NUOP*U2K)                              // 4352
#define UPACK   (USEG1 + USEG2 + USEG3)                 // 77824

#define AS1 __attribute__((address_space(1)))
#define AS3 __attribute__((address_space(3)))

#if defined(__gfx1250__) && __has_builtin(__builtin_amdgcn_global_load_async_to_lds_b128)
#define HAVE_ASYNC_LDS 1
#else
#define HAVE_ASYNC_LDS 0
#endif

__device__ __forceinline__ void cp16_g2l(const __bf16* g, __bf16* l) {
#if HAVE_ASYNC_LDS
  __builtin_amdgcn_global_load_async_to_lds_b128(
      (AS1 v4i_g*)(__bf16*)g, (AS3 v4i_g*)l, 0, 0);
#else
  *(v4u*)l = *(const v4u*)g;
#endif
}

__device__ __forceinline__ void wait_async_lds() {
#if HAVE_ASYNC_LDS
#if __has_builtin(__builtin_amdgcn_s_wait_asynccnt)
  __builtin_amdgcn_s_wait_asynccnt(0);
#else
  asm volatile("s_wait_asynccnt 0" ::: "memory");
#endif
#endif
}

// copy n16 16-byte granules global->LDS, block-cooperative (256 threads)
__device__ __forceinline__ void copy_seg(const __bf16* g, __bf16* l, int n16,
                                         int tid) {
  for (int i = tid; i < n16; i += 256) cp16_g2l(g + i * 8, l + i * 8);
}

__device__ __forceinline__ float elu_f(float v) {
  return v > 0.0f ? v : (__expf(v) - 1.0f);
}

__device__ __forceinline__ v8f zero8() {
  v8f z;
#pragma unroll
  for (int i = 0; i < 8; ++i) z[i] = 0.0f;
  return z;
}

__device__ __forceinline__ v8f wmma_bf16(v16bf a, v16bf b, v8f c) {
  // D(f32 16x16) = A(bf16 16x32) * B(bf16 32x16) + C
  return __builtin_amdgcn_wmma_f32_16x16x32_bf16(false, a, false, b, (short)0, c,
                                                 false, false);
}

// A operand (bf16 16x32) from an f32 row in global memory.
// A k order used here (contiguous): lanes 0-15 hold K = 0..15, lanes 16-31 K = 16..31 (matches B).
__device__ __forceinline__ v16bf make_a_f32(const float* rowp, int k0, int half) {
  const float* c0 = rowp + k0 + half * 16;
  const float* c1 = rowp + k0 + half * 16 + 8;
  v4f a0 = *(const v4f*)(c0);
  v4f a1 = *(const v4f*)(c0 + 4);
  v4f b0 = *(const v4f*)(c1);
  v4f b1 = *(const v4f*)(c1 + 4);
  v16bf r;
#pragma unroll
  for (int i = 0; i < 4; ++i) {
    r[i]      = (__bf16)a0[i];
    r[4 + i]  = (__bf16)a1[i];
    r[8 + i]  = (__bf16)b0[i];
    r[12 + i] = (__bf16)b1[i];
  }
  return r;
}

// A operand from a bf16 row staged in LDS.
__device__ __forceinline__ v16bf make_a_bf16(const __bf16* rowp, int k0, int half) {
  v8bf c0 = *(const v8bf*)(rowp + k0 + half * 16);
  v8bf c1 = *(const v8bf*)(rowp + k0 + half * 16 + 8);
  v16bf r;
#pragma unroll
  for (int i = 0; i < 8; ++i) { r[i] = c0[i]; r[8 + i] = c1[i]; }
  return r;
}

// B operand: 16 contiguous bf16 (lanes 0-15 read K=k0.., lanes 16-31 K=k0+16..).
__device__ __forceinline__ v16bf load16_bf16(const __bf16* p) {
  v8bf c0 = *(const v8bf*)(p);
  v8bf c1 = *(const v8bf*)(p + 8);
  v16bf r;
#pragma unroll
  for (int i = 0; i < 8; ++i) { r[i] = c0[i]; r[8 + i] = c1[i]; }
  return r;
}

// ---------------------------------------------------------------------------
// Pack kernels: build the bf16 transposed/zero-padded LDS images once.
// ---------------------------------------------------------------------------
__global__ __launch_bounds__(256) void pack_heads_kernel(
    const float* __restrict__ W1, const float* __restrict__ W2,
    const float* __restrict__ W3, __bf16* __restrict__ pw) {
  const int g = blockIdx.y;
  const int i = blockIdx.x * 256 + threadIdx.x;
  if (i >= GPACK) return;
  float v = 0.0f;
  int r = i;
  if (r < HSEG1) {
    int n = r / W1K, k = r % W1K;
    if (n < NH && k < NF) v = W1[(g * NF + k) * NH + n];
  } else if ((r -= HSEG1) < HSEG2) {
    int n = r / W2K, k = r % W2K;
    if (n < NH && k < NH) v = W2[(g * NH + k) * NH + n];
  } else {
    r -= HSEG2;
    int n = r / W2K, k = r % W2K;
    if (n < NO && k < NH) v = W3[(g * NH + k) * NO + n];
  }
  pw[(size_t)g * GPACK + i] = (__bf16)v;
}

__global__ __launch_bounds__(256) void pack_univ_kernel(
    const float* __restrict__ uW1, const float* __restrict__ uW2,
    const float* __restrict__ uW3, __bf16* __restrict__ pwu) {
  const int i = blockIdx.x * 256 + threadIdx.x;
  if (i >= UPACK) return;
  float v = 0.0f;
  int r = i;
  if (r < USEG1) {
    int n = r / U1K, k = r % U1K;
    if (n < NUH && k < NUF) v = uW1[k * NUH + n];
  } else if ((r -= USEG1) < USEG2) {
    int n = r / U2K, k = r % U2K;
    if (n < NUH && k < NUH) v = uW2[k * NUH + n];
  } else {
    r -= USEG2;
    int n = r / U2K, k = r % U2K;
    if (n < NUO && k < NUH) v = uW3[k * NUO + n];
  }
  pwu[i] = (__bf16)v;
}

// ---------------------------------------------------------------------------
// Kernel 1: 32 group heads. grid = (B/128, G), block = 256 (8 waves x 16 rows)
// ---------------------------------------------------------------------------
__global__ __launch_bounds__(256) void heads_kernel(
    const float* __restrict__ x, const __bf16* __restrict__ pw,
    const float* __restrict__ b1, const float* __restrict__ b2,
    const float* __restrict__ b3, float* __restrict__ hws) {
  __shared__ alignas(16) __bf16 sW1[NHP][W1K];   // W1^T, zero-padded 52->64
  __shared__ alignas(16) __bf16 sW2[NHP][W2K];   // W2^T
  __shared__ alignas(16) __bf16 sW3[NOP][W2K];   // W3^T, 5->16
  __shared__ float sb1[NHP];
  __shared__ float sb2[NHP];
  __shared__ float sb3[NOP];
  __shared__ alignas(16) __bf16 sh[8][16][W2K];  // per-wave activation staging

  const int g   = blockIdx.y;
  const int tid = threadIdx.x;

  // async bulk copy of the packed bf16 weight image into LDS
  const __bf16* wp = pw + (size_t)g * GPACK;
  copy_seg(wp,                 &sW1[0][0], HSEG1 / 8, tid);
  copy_seg(wp + HSEG1,         &sW2[0][0], HSEG2 / 8, tid);
  copy_seg(wp + HSEG1 + HSEG2, &sW3[0][0], HSEG3 / 8, tid);
  if (tid < NHP) {
    sb1[tid] = (tid < NH) ? b1[g * NH + tid] : 0.0f;
    sb2[tid] = (tid < NH) ? b2[g * NH + tid] : 0.0f;
  }
  if (tid < NOP) sb3[tid] = (tid < NO) ? b3[g * NO + tid] : 0.0f;
  wait_async_lds();
  __syncthreads();

  const int wave = tid >> 5;
  const int lane = tid & 31;
  const int m    = lane & 15;
  const int half = lane >> 4;
  const int row0 = blockIdx.x * 128 + wave * 16;
  const float* xrow = x + (size_t)(row0 + m) * XSTRIDE + g * NF;

  // ---- layer 1: [16x256] x [256x64] ----
  v8f acc[4];
#pragma unroll
  for (int nt = 0; nt < 4; ++nt) acc[nt] = zero8();
#pragma unroll
  for (int k0 = 0; k0 < NF; k0 += 32) {
    v16bf a = make_a_f32(xrow, k0, half);
#pragma unroll
    for (int nt = 0; nt < 4; ++nt) {
      v16bf b = load16_bf16(&sW1[nt * 16 + m][k0 + half * 16]);
      acc[nt] = wmma_bf16(a, b, acc[nt]);
    }
  }
  // bias + ELU, restage through LDS (D is column-per-lane; A needs row-per-lane)
#pragma unroll
  for (int nt = 0; nt < 4; ++nt) {
    int col = nt * 16 + m;
    float bias = sb1[col];
#pragma unroll
    for (int r = 0; r < 8; ++r) {
      float v = elu_f(acc[nt][r] + bias);
      sh[wave][r + half * 8][col] = (__bf16)v;
    }
  }

  // ---- layer 2: [16x64] x [64x64] ----
  v8f acc2[4];
#pragma unroll
  for (int nt = 0; nt < 4; ++nt) acc2[nt] = zero8();
#pragma unroll
  for (int k0 = 0; k0 < NHP; k0 += 32) {
    v16bf a = make_a_bf16(&sh[wave][m][0], k0, half);
#pragma unroll
    for (int nt = 0; nt < 4; ++nt) {
      v16bf b = load16_bf16(&sW2[nt * 16 + m][k0 + half * 16]);
      acc2[nt] = wmma_bf16(a, b, acc2[nt]);
    }
  }
#pragma unroll
  for (int nt = 0; nt < 4; ++nt) {
    int col = nt * 16 + m;
    float bias = sb2[col];
#pragma unroll
    for (int r = 0; r < 8; ++r) {
      float v = elu_f(acc2[nt][r] + bias);
      sh[wave][r + half * 8][col] = (__bf16)v;
    }
  }

  // ---- layer 3: [16x64] x [64x16] ----
  v8f acc3 = zero8();
#pragma unroll
  for (int k0 = 0; k0 < NHP; k0 += 32) {
    v16bf a = make_a_bf16(&sh[wave][m][0], k0, half);
    v16bf b = load16_bf16(&sW3[m][k0 + half * 16]);
    acc3 = wmma_bf16(a, b, acc3);
  }
  if (m < NO) {
    float bias = sb3[m];
#pragma unroll
    for (int r = 0; r < 8; ++r) {
      float v = elu_f(acc3[r] + bias);
      hws[(size_t)(row0 + r + half * 8) * OUTW_N + g * NO + m] = v;
    }
  }
}

// ---------------------------------------------------------------------------
// Kernel 2: universal MLP 512 -> 112(pad103) -> 112 -> 32(pad20)
// ---------------------------------------------------------------------------
__global__ __launch_bounds__(256) void universal_kernel(
    const float* __restrict__ x, const __bf16* __restrict__ pwu,
    const float* __restrict__ ub1, const float* __restrict__ ub2,
    const float* __restrict__ ub3, float* __restrict__ hws) {
  __shared__ alignas(16) __bf16 sU1[NUHP][U1K];
  __shared__ alignas(16) __bf16 sU2[NUHP][U2K];
  __shared__ alignas(16) __bf16 sU3[NUOP][U2K];
  __shared__ float sb1[NUHP];
  __shared__ float sb2[NUHP];
  __shared__ float sb3[NUOP];
  __shared__ alignas(16) __bf16 sh[8][16][U2K];

  const int tid = threadIdx.x;
  copy_seg(pwu,                 &sU1[0][0], USEG1 / 8, tid);
  copy_seg(pwu + USEG1,         &sU2[0][0], USEG2 / 8, tid);
  copy_seg(pwu + USEG1 + USEG2, &sU3[0][0], USEG3 / 8, tid);
  if (tid < NUHP) {
    sb1[tid] = (tid < NUH) ? ub1[tid] : 0.0f;
    sb2[tid] = (tid < NUH) ? ub2[tid] : 0.0f;
  }
  if (tid < NUOP) sb3[tid] = (tid < NUO) ? ub3[tid] : 0.0f;
  wait_async_lds();
  __syncthreads();

  const int wave = tid >> 5;
  const int lane = tid & 31;
  const int m    = lane & 15;
  const int half = lane >> 4;
  const int row0 = blockIdx.x * 128 + wave * 16;
  const float* xrow = x + (size_t)(row0 + m) * XSTRIDE + NG * NF;

  // zero the K-pad columns (112..127) of this wave's staging tile
#pragma unroll
  for (int r = 0; r < 8; ++r) sh[wave][r + half * 8][NUHP + m] = (__bf16)0.0f;

  // ---- layer 1: [16x512] x [512x112] ----
  v8f acc[7];
#pragma unroll
  for (int nt = 0; nt < 7; ++nt) acc[nt] = zero8();
  for (int k0 = 0; k0 < NUF; k0 += 32) {
    v16bf a = make_a_f32(xrow, k0, half);
#pragma unroll
    for (int nt = 0; nt < 7; ++nt) {
      v16bf b = load16_bf16(&sU1[nt * 16 + m][k0 + half * 16]);
      acc[nt] = wmma_bf16(a, b, acc[nt]);
    }
  }
#pragma unroll
  for (int nt = 0; nt < 7; ++nt) {
    int col = nt * 16 + m;
    float bias = sb1[col];
#pragma unroll
    for (int r = 0; r < 8; ++r) {
      float v = elu_f(acc[nt][r] + bias);
      sh[wave][r + half * 8][col] = (__bf16)v;
    }
  }

  // ---- layer 2: [16x128] x [128x112] ----
  v8f acc2[7];
#pragma unroll
  for (int nt = 0; nt < 7; ++nt) acc2[nt] = zero8();
#pragma unroll
  for (int k0 = 0; k0 < NK2P; k0 += 32) {
    v16bf a = make_a_bf16(&sh[wave][m][0], k0, half);
#pragma unroll
    for (int nt = 0; nt < 7; ++nt) {
      v16bf b = load16_bf16(&sU2[nt * 16 + m][k0 + half * 16]);
      acc2[nt] = wmma_bf16(a, b, acc2[nt]);
    }
  }
#pragma unroll
  for (int nt = 0; nt < 7; ++nt) {
    int col = nt * 16 + m;
    float bias = sb2[col];
#pragma unroll
    for (int r = 0; r < 8; ++r) {
      float v = elu_f(acc2[nt][r] + bias);
      sh[wave][r + half * 8][col] = (__bf16)v;
    }
  }

  // ---- layer 3: [16x128] x [128x32] ----
  v8f acc3[2];
#pragma unroll
  for (int nt = 0; nt < 2; ++nt) acc3[nt] = zero8();
#pragma unroll
  for (int k0 = 0; k0 < NK2P; k0 += 32) {
    v16bf a = make_a_bf16(&sh[wave][m][0], k0, half);
#pragma unroll
    for (int nt = 0; nt < 2; ++nt) {
      v16bf b = load16_bf16(&sU3[nt * 16 + m][k0 + half * 16]);
      acc3[nt] = wmma_bf16(a, b, acc3[nt]);
    }
  }
#pragma unroll
  for (int nt = 0; nt < 2; ++nt) {
    int col = nt * 16 + m;
    if (col < NUO) {
      float bias = sb3[col];
#pragma unroll
      for (int r = 0; r < 8; ++r) {
        float v = elu_f(acc3[nt][r] + bias);
        hws[(size_t)(row0 + r + half * 8) * OUTW_N + NG * NO + col] = v;
      }
    }
  }
}

// ---------------------------------------------------------------------------
// Kernel 3: out[b] = sum_j hws[b][j] * outW[j]
// ---------------------------------------------------------------------------
__global__ __launch_bounds__(256) void reduce_kernel(
    const float* __restrict__ hws, const float* __restrict__ outW,
    float* __restrict__ out) {
  __shared__ float sw[OUTW_N];
  const int tid = threadIdx.x;
  if (tid < OUTW_N) sw[tid] = outW[tid];
  __syncthreads();
  const int b = blockIdx.x * 256 + tid;
  const float* p = hws + (size_t)b * OUTW_N;
  float acc = 0.0f;
#pragma unroll 4
  for (int j = 0; j < OUTW_N; ++j) acc += p[j] * sw[j];
  out[b] = acc;
}

extern "C" void kernel_launch(void* const* d_in, const int* in_sizes, int n_in,
                              void* d_out, int out_size, void* d_ws, size_t ws_size,
                              hipStream_t stream) {
  (void)in_sizes; (void)n_in; (void)out_size; (void)ws_size;
  const float* x    = (const float*)d_in[0];
  const float* W1   = (const float*)d_in[1];
  const float* b1   = (const float*)d_in[2];
  const float* W2   = (const float*)d_in[3];
  const float* b2   = (const float*)d_in[4];
  const float* W3   = (const float*)d_in[5];
  const float* b3   = (const float*)d_in[6];
  const float* uW1  = (const float*)d_in[7];
  const float* ub1  = (const float*)d_in[8];
  const float* uW2  = (const float*)d_in[9];
  const float* ub2  = (const float*)d_in[10];
  const float* uW3  = (const float*)d_in[11];
  const float* ub3  = (const float*)d_in[12];
  const float* outW = (const float*)d_in[13];
  float*  hws = (float*)d_ws;                 // B x 180 f32 staging
  __bf16* pw  = (__bf16*)(hws + HWS_ELEMS);   // packed bf16 weight images
  __bf16* pwu = pw + HPACK_TOT;
  float*  out = (float*)d_out;

  dim3 gridPH((GPACK + 255) / 256, NG);
  pack_heads_kernel<<<gridPH, 256, 0, stream>>>(W1, W2, W3, pw);
  pack_univ_kernel<<<(UPACK + 255) / 256, 256, 0, stream>>>(uW1, uW2, uW3, pwu);

  dim3 gridH(NB / 128, NG);
  heads_kernel<<<gridH, 256, 0, stream>>>(x, pw, b1, b2, b3, hws);
  universal_kernel<<<NB / 128, 256, 0, stream>>>(x, pwu, ub1, ub2, ub3, hws);
  reduce_kernel<<<NB / 256, 256, 0, stream>>>(hws, outW, out);
}
